// BatchedSpGat_62964220560017
// MI455X (gfx1250) — matrix operand order repaired and checked by reading, not compiled, not run on hardware
//
#include <hip/hip_runtime.h>
#include <hip/hip_bf16.h>
#include <math.h>

// ---------------- problem constants ----------------
#define NNODES 50000
#define NEDGES 800000
#define NTOT   (NEDGES + NNODES)   // 850000 edges incl. self loops
#define FIN    512
#define F1     128                 // layer1: H=4, C=32
#define F2     256                 // layer2: H=1, C=256

// ---------------- CDNA5 WMMA types ----------------
typedef __attribute__((ext_vector_type(16))) __bf16 v16bf;
typedef __attribute__((ext_vector_type(8)))  float  v8f;

__device__ __forceinline__ unsigned short f2bf(float f) {
  unsigned x = __float_as_uint(f);
  unsigned r = x + 0x7fffu + ((x >> 16) & 1u);   // RNE
  return (unsigned short)(r >> 16);
}

// order-preserving float <-> uint map (exact float atomicMax via uint atomicMax)
__device__ __forceinline__ unsigned fmap(float f) {
  unsigned u = __float_as_uint(f);
  return (u & 0x80000000u) ? ~u : (u | 0x80000000u);
}
__device__ __forceinline__ float funmap(unsigned u) {
  unsigned b = (u & 0x80000000u) ? (u ^ 0x80000000u) : ~u;
  return __uint_as_float(b);
}

__device__ __forceinline__ void edge_sd(const int* __restrict__ ei, int e,
                                        int& s, int& d) {
  if (e < NEDGES) { s = ei[e]; d = ei[NEDGES + e]; }
  else            { s = d = e - NEDGES; }          // self loop
}

// =====================================================================
// C[M,N] = A[M,K] @ B[K,N]  (fp32 in/out, bf16 WMMA compute, f32 accum)
// block = 256 threads (8 waves); tile 64(M) x 128(N); K-step 32.
// LDS holds tiles PRE-SWIZZLED into WMMA fragment order so each lane's
// 16 bf16 operand values are one contiguous 32B run (2x ds_load_b128).
//
//  A frag layout (16-bit A 16x32): lanes 0-15: row=lane,   K=0..15 sequential
//                                  lanes16-31: row=lane-16,K=16..31 sequential
//   -> Af[row][group][16]  where group=k>>4, idx=k&15
//  B frag layout (32x16): lanes 0-15: col=lane,   K=0..15 sequential
//                         lanes16-31: col=lane-16,K=16..31 sequential
//   -> Bf[j][lane][16]     where j=n>>4, lane=(n&15)+((k>>4)<<4), idx=k&15
// Requires: N % 128 == 0, K % 32 == 0 (true here: N=128/256, K=512/128).
// =====================================================================
__global__ __launch_bounds__(256) void gemm_bf16(const float* __restrict__ A,
                                                 const float* __restrict__ B,
                                                 float* __restrict__ C,
                                                 int M, int N, int K) {
  __shared__ unsigned short Af[64 * 2 * 16];   // 4 KB, fragment-ready A
  __shared__ unsigned short Bf[4 * 32 * 16];   // 4 KB, fragment-ready B

  const int t    = threadIdx.x;
  const int lane = t & 31;
  const int wave = t >> 5;
  const int wr   = wave & 3;          // row tile (16 rows)
  const int wc   = wave >> 2;         // col half (64 cols)
  const int rowBase = blockIdx.x * 64;
  const int colBase = blockIdx.y * 128;

  v8f acc[4];
#pragma unroll
  for (int j = 0; j < 4; ++j) acc[j] = (v8f){0,0,0,0,0,0,0,0};

  for (int k0 = 0; k0 < K; k0 += 32) {
    // ---- stage A tile 64x32: thread -> (row, 8-k chunk); one 16B LDS store
    {
      int r  = t >> 2;                // 0..63
      int c  = (t & 3) * 8;           // 0,8,16,24
      int gr = rowBase + r;
      unsigned short tmp[8];
      if (gr < M) {
        const float4* p = reinterpret_cast<const float4*>(A + (size_t)gr * K + k0 + c);
        float4 v0 = p[0], v1 = p[1];
        tmp[0] = f2bf(v0.x); tmp[1] = f2bf(v0.y); tmp[2] = f2bf(v0.z); tmp[3] = f2bf(v0.w);
        tmp[4] = f2bf(v1.x); tmp[5] = f2bf(v1.y); tmp[6] = f2bf(v1.z); tmp[7] = f2bf(v1.w);
      } else {
#pragma unroll
        for (int i = 0; i < 8; ++i) tmp[i] = 0;
      }
      int group = c >> 4;             // c=0->g0i0, 8->g0i8, 16->g1i0, 24->g1i8
      int idxb  = ((c >> 3) & 1) << 3;
      *reinterpret_cast<uint4*>(&Af[(r * 2 + group) * 16 + idxb]) =
          *reinterpret_cast<uint4*>(tmp);
    }
    // ---- stage B tile 32x128: thread -> (col n, 16-deep k column);
    //      global reads stay coalesced across threads (row segments),
    //      one contiguous 32B LDS store at the fragment address.
    {
      int n  = t & 127;               // 0..127
      int kg = t >> 7;                // 0..1  (k = kg*16 .. +15)
      const float* bp = B + (size_t)(k0 + kg * 16) * N + colBase + n;
      unsigned short tmp[16];
#pragma unroll
      for (int i = 0; i < 16; ++i) tmp[i] = f2bf(bp[(size_t)i * N]);
      int j  = n >> 4;
      int ln = (n & 15) + (kg << 4);
      uint4* dst = reinterpret_cast<uint4*>(&Bf[(j * 32 + ln) * 16]);
      dst[0] = *reinterpret_cast<uint4*>(&tmp[0]);
      dst[1] = *reinterpret_cast<uint4*>(&tmp[8]);
    }
    __syncthreads();

    // ---- fragments: one 32B contiguous LDS read each
    v16bf av = *reinterpret_cast<const v16bf*>(
        &Af[((wr * 16 + (lane & 15)) * 2 + (lane >> 4)) * 16]);
#pragma unroll
    for (int j = 0; j < 4; ++j) {
      int jj = wc * 4 + j;            // global 16-col tile index 0..7
      v16bf bv = *reinterpret_cast<const v16bf*>(&Bf[(jj * 32 + lane) * 16]);
      acc[j] = __builtin_amdgcn_wmma_f32_16x16x32_bf16(
          /*neg_a=*/false, av, /*neg_b=*/false, bv,
          /*c_mod=*/(short)0, acc[j], /*reuse_a=*/false, /*reuse_b=*/false);
    }
    __syncthreads();
  }

  // ---- store (f32 C/D layout: VGPR r -> M=r (lanes 0-15) / M=8+r (16-31))
#pragma unroll
  for (int j = 0; j < 4; ++j) {
    int col = colBase + wc * 64 + j * 16 + (lane & 15);
#pragma unroll
    for (int r = 0; r < 8; ++r) {
      int row = rowBase + wr * 16 + r + ((lane >> 4) * 8);
      if (row < M) C[(size_t)row * N + col] = acc[j][r];
    }
  }
}

// =====================================================================
// per-(node,head) attention dots: al_s[n,h] = <h[n,h,:], a_s[h,:]>, same a_d
// one wave per (node,head)
// =====================================================================
__global__ __launch_bounds__(256) void att_logits(const float* __restrict__ h,
                                                  const float* __restrict__ as,
                                                  const float* __restrict__ ad,
                                                  float* __restrict__ als,
                                                  float* __restrict__ ald,
                                                  int H, int C) {
  int lane = threadIdx.x & 31;
  int wid  = blockIdx.x * (blockDim.x >> 5) + (threadIdx.x >> 5);
  if (wid >= NNODES * H) return;
  int hd = wid % H;
  const float* hp = h + (size_t)wid * C;   // h is [N, H, C] flat; wid = n*H+hd
  float ss = 0.f, sd = 0.f;
  for (int c = lane; c < C; c += 32) {
    float v = hp[c];
    ss += v * as[hd * C + c];
    sd += v * ad[hd * C + c];
  }
#pragma unroll
  for (int m = 16; m; m >>= 1) {
    ss += __shfl_xor(ss, m, 32);
    sd += __shfl_xor(sd, m, 32);
  }
  if (lane == 0) { als[wid] = ss; ald[wid] = sd; }
}

// =====================================================================
// pass 1: logits = leaky_relu(al_s[src]+al_d[dst]); store + segment max
// =====================================================================
__global__ __launch_bounds__(256) void edge_max(const int* __restrict__ ei,
                                                const float* __restrict__ als,
                                                const float* __restrict__ ald,
                                                float* __restrict__ lg,
                                                unsigned* __restrict__ mb, int H) {
  int t = blockIdx.x * blockDim.x + threadIdx.x;
  if (t >= NTOT * H) return;
  int e = t / H, hd = t - e * H;
  int s, d; edge_sd(ei, e, s, d);
  float l = als[s * H + hd] + ald[d * H + hd];
  l = (l > 0.f) ? l : 0.2f * l;             // leaky relu, slope 0.2
  lg[t] = l;
  atomicMax(&mb[d * H + hd], fmap(l));
}

// pass 2: e = exp(l - m[dst]); overwrite lg; segment sum via HW f32 atomic
__global__ __launch_bounds__(256) void edge_expsum(const int* __restrict__ ei,
                                                   float* __restrict__ lg,
                                                   const unsigned* __restrict__ mb,
                                                   float* __restrict__ sb, int H) {
  int t = blockIdx.x * blockDim.x + threadIdx.x;
  if (t >= NTOT * H) return;
  int e = t / H, hd = t - e * H;
  int s, d; edge_sd(ei, e, s, d);
  (void)s;
  float ex = __expf(lg[t] - funmap(mb[d * H + hd]));
  lg[t] = ex;
  __hip_atomic_fetch_add(&sb[d * H + hd], ex, __ATOMIC_RELAXED, __HIP_MEMORY_SCOPE_AGENT);
}

// pass 3: out[dst] += (e/s[dst]) * h[src] -- one wave per edge, float4 lanes
// F = H<<cshift total channels (128 or 256); 128 channels per chunk.
__global__ __launch_bounds__(256) void edge_aggr(const int* __restrict__ ei,
                                                 const float* __restrict__ ew,
                                                 const float* __restrict__ sb,
                                                 const float* __restrict__ h,
                                                 float* __restrict__ out,
                                                 int H, int cshift) {
  int lane = threadIdx.x & 31;
  int w = blockIdx.x * (blockDim.x >> 5) + (threadIdx.x >> 5);
  if (w >= NTOT) return;
  int s, d; edge_sd(ei, w, s, d);
  int F = H << cshift;
  int chunks = F >> 7;
  for (int ch = 0; ch < chunks; ++ch) {
    int c  = ch * 128 + lane * 4;
    int hd = c >> cshift;
    float alpha = ew[(size_t)w * H + hd] / sb[(size_t)d * H + hd];
    float4 hv = *reinterpret_cast<const float4*>(h + (size_t)s * F + c);
    float* op = out + (size_t)d * F + c;
    __hip_atomic_fetch_add(op + 0, alpha * hv.x, __ATOMIC_RELAXED, __HIP_MEMORY_SCOPE_AGENT);
    __hip_atomic_fetch_add(op + 1, alpha * hv.y, __ATOMIC_RELAXED, __HIP_MEMORY_SCOPE_AGENT);
    __hip_atomic_fetch_add(op + 2, alpha * hv.z, __ATOMIC_RELAXED, __HIP_MEMORY_SCOPE_AGENT);
    __hip_atomic_fetch_add(op + 3, alpha * hv.w, __ATOMIC_RELAXED, __HIP_MEMORY_SCOPE_AGENT);
  }
}

// init helpers ---------------------------------------------------------
__global__ void fill_bias(float* __restrict__ o, const float* __restrict__ b,
                          int total, int Fmask) {
  for (int i = blockIdx.x * blockDim.x + threadIdx.x; i < total;
       i += gridDim.x * blockDim.x)
    o[i] = b[i & Fmask];
}
__global__ void fill_zero(unsigned* __restrict__ p, int n) {
  for (int i = blockIdx.x * blockDim.x + threadIdx.x; i < n;
       i += gridDim.x * blockDim.x)
    p[i] = 0u;
}

// final: y = relu(out2 / max(||out2||_2, 1e-12)) -- one wave per node
__global__ __launch_bounds__(256) void finalize_norm(const float* __restrict__ o2,
                                                     float* __restrict__ y) {
  int lane = threadIdx.x & 31;
  int n = blockIdx.x * (blockDim.x >> 5) + (threadIdx.x >> 5);
  if (n >= NNODES) return;
  const float4* p = reinterpret_cast<const float4*>(o2 + (size_t)n * F2);
  float4 v0 = p[lane], v1 = p[32 + lane];
  float ss = v0.x*v0.x + v0.y*v0.y + v0.z*v0.z + v0.w*v0.w
           + v1.x*v1.x + v1.y*v1.y + v1.z*v1.z + v1.w*v1.w;
#pragma unroll
  for (int m = 16; m; m >>= 1) ss += __shfl_xor(ss, m, 32);
  float inv = 1.0f / fmaxf(sqrtf(ss), 1e-12f);
  float4* q = reinterpret_cast<float4*>(y + (size_t)n * F2);
  float4 r0, r1;
  r0.x = fmaxf(v0.x * inv, 0.f); r0.y = fmaxf(v0.y * inv, 0.f);
  r0.z = fmaxf(v0.z * inv, 0.f); r0.w = fmaxf(v0.w * inv, 0.f);
  r1.x = fmaxf(v1.x * inv, 0.f); r1.y = fmaxf(v1.y * inv, 0.f);
  r1.z = fmaxf(v1.z * inv, 0.f); r1.w = fmaxf(v1.w * inv, 0.f);
  q[lane] = r0; q[32 + lane] = r1;
}

// =====================================================================
extern "C" void kernel_launch(void* const* d_in, const int* in_sizes, int n_in,
                              void* d_out, int out_size, void* d_ws, size_t ws_size,
                              hipStream_t stream) {
  const float* x   = (const float*)d_in[0];
  const int*   ei  = (const int*)  d_in[1];   // [2, E], src row then dst row
  const float* W1  = (const float*)d_in[2];
  const float* a1s = (const float*)d_in[3];
  const float* a1d = (const float*)d_in[4];
  const float* b1  = (const float*)d_in[5];
  const float* W2  = (const float*)d_in[6];
  const float* a2s = (const float*)d_in[7];
  const float* a2d = (const float*)d_in[8];
  const float* b2  = (const float*)d_in[9];
  float* y = (float*)d_out;

  // workspace layout (with aliasing; ~119 MB)
  float*    h1   = (float*)d_ws;                       // [N,128]
  float*    out1 = h1   + (size_t)NNODES * F1;         // [N,128]
  float*    h2   = out1 + (size_t)NNODES * F1;         // [N,256]
  float*    al1s = h2   + (size_t)NNODES * F2;         // [N,4]
  float*    al1d = al1s + (size_t)NNODES * 4;          // [N,4]
  unsigned* m1   = (unsigned*)(al1d + (size_t)NNODES * 4); // [N,4]
  float*    s1   = (float*)(m1 + (size_t)NNODES * 4);  // [N,4]
  float*    lg1  = s1 + (size_t)NNODES * 4;            // [T,4]
  // layer-2 aliases (layer-1 state dead by then)
  float*    out2 = h1;          // [N,256] == h1 ∪ out1 (contiguous)
  float*    al2s = al1s;  float* al2d = al1d;
  unsigned* m2   = m1;    float* s2   = s1;
  float*    lg2  = lg1;

  const dim3 blk(256);
  auto cdiv = [](int a, int b) { return (a + b - 1) / b; };

  // ---------------- layer 1 (H=4, C=32) ----------------
  fill_zero<<<cdiv(NNODES * 8, 256), blk, 0, stream>>>(m1, NNODES * 8); // m1+s1
  fill_bias<<<2048, blk, 0, stream>>>(out1, b1, NNODES * F1, F1 - 1);

  gemm_bf16<<<dim3(cdiv(NNODES, 64), F1 / 128), blk, 0, stream>>>(
      x, W1, h1, NNODES, F1, FIN);

  att_logits<<<cdiv(NNODES * 4, 8), blk, 0, stream>>>(h1, a1s, a1d, al1s, al1d, 4, 32);
  edge_max   <<<cdiv(NTOT * 4, 256), blk, 0, stream>>>(ei, al1s, al1d, lg1, m1, 4);
  edge_expsum<<<cdiv(NTOT * 4, 256), blk, 0, stream>>>(ei, lg1, m1, s1, 4);
  edge_aggr  <<<cdiv(NTOT, 8),       blk, 0, stream>>>(ei, lg1, s1, h1, out1, 4, 5);

  // ---------------- layer 2 (H=1, C=256) ----------------
  gemm_bf16<<<dim3(cdiv(NNODES, 64), F2 / 128), blk, 0, stream>>>(
      out1, W2, h2, NNODES, F2, F1);

  // out2 aliases h1∪out1: must init only after gemm2 consumed out1 (stream order)
  fill_zero<<<cdiv(NNODES, 256), blk, 0, stream>>>(m2, NNODES);
  fill_zero<<<cdiv(NNODES, 256), blk, 0, stream>>>((unsigned*)s2, NNODES);
  fill_bias<<<4096, blk, 0, stream>>>(out2, b2, NNODES * F2, F2 - 1);

  att_logits<<<cdiv(NNODES, 8),  blk, 0, stream>>>(h2, a2s, a2d, al2s, al2d, 1, 256);
  edge_max   <<<cdiv(NTOT, 256), blk, 0, stream>>>(ei, al2s, al2d, lg2, m2, 1);
  edge_expsum<<<cdiv(NTOT, 256), blk, 0, stream>>>(ei, lg2, m2, s2, 1);
  edge_aggr  <<<cdiv(NTOT, 8),   blk, 0, stream>>>(ei, lg2, s2, h2, out2, 1, 8);

  finalize_norm<<<cdiv(NNODES, 8), blk, 0, stream>>>(out2, y);
}
